// ImageSelfAttention_42047729828325
// MI455X (gfx1250) — compile-verified, matrix operand order checked by reading, not run on hardware
//
#include <hip/hip_runtime.h>
#include <hip/hip_bf16.h>

typedef __attribute__((ext_vector_type(16))) __bf16 bf16x16;
typedef __attribute__((ext_vector_type(8)))  float  f32x8;
typedef __attribute__((ext_vector_type(4)))  int    v4i;

#if defined(__gfx1250__) && __has_builtin(__builtin_amdgcn_global_load_async_to_lds_b128) && __has_builtin(__builtin_amdgcn_s_wait_asynccnt)
#define USE_ASYNC_LDS 1
#else
#define USE_ASYNC_LDS 0
#endif

// ---------------------------------------------------------------------------
// Kernel 0: f32 -> bf16 conversion WITH transpose.
// src is (rows x cols) row-major; dst[c * rows + r] = src[r * cols + c].
// For wq/wk/wv: rows=4096(K), cols=512(N)  -> WT[n][k], K contiguous.
// For wo:       rows=512(K), cols=4096(N)  -> WoT[n][k], K contiguous.
// ---------------------------------------------------------------------------
__global__ void cvt_bf16_t_kernel(const float* __restrict__ src,
                                  __bf16* __restrict__ dst, int rows, int cols) {
    int idx = blockIdx.x * blockDim.x + threadIdx.x;
    if (idx < rows * cols) {
        int r = idx / cols, c = idx % cols;
        dst[(size_t)c * rows + r] = (__bf16)src[idx];
    }
}

// ---------------------------------------------------------------------------
// Kernel 1: patchify + LayerNorm -> Y (8192 x 4096) bf16
// ---------------------------------------------------------------------------
__global__ __launch_bounds__(256)
void patchify_ln_kernel(const float* __restrict__ x,
                        const float* __restrict__ gamma,
                        const float* __restrict__ beta,
                        __bf16* __restrict__ Y) {
    const int token = blockIdx.x;          // 0..8191
    const int b   = token >> 8;
    const int tok = token & 255;
    const int pi  = tok >> 4;              // i
    const int pj  = tok & 15;              // j
    const int tid = threadIdx.x;

    float v[16];
    float s = 0.f, ss = 0.f;
#pragma unroll
    for (int ii = 0; ii < 16; ++ii) {
        int fr = ii * 256 + tid;
        int pr = fr >> 9;
        int pc = (fr >> 6) & 7;
        int cc = fr & 63;
        size_t xi = ((size_t)(b * 128 + (pj * 8 + pr)) * 128 + (pi * 8 + pc)) * 64 + cc;
        float t = x[xi];
        v[ii] = t;
        s  += t;
        ss += t * t;
    }
#pragma unroll
    for (int off = 16; off >= 1; off >>= 1) {
        s  += __shfl_xor(s, off, 32);
        ss += __shfl_xor(ss, off, 32);
    }
    __shared__ float red[2][8];
    const int wave = tid >> 5, lane = tid & 31;
    if (lane == 0) { red[0][wave] = s; red[1][wave] = ss; }
    __syncthreads();
    if (tid == 0) {
        float a = 0.f, c = 0.f;
        for (int w = 0; w < 8; ++w) { a += red[0][w]; c += red[1][w]; }
        red[0][0] = a; red[1][0] = c;
    }
    __syncthreads();
    const float mean = red[0][0] * (1.0f / 4096.0f);
    const float var  = red[1][0] * (1.0f / 4096.0f) - mean * mean;
    const float rstd = rsqrtf(var + 1e-6f);

    __bf16* out = Y + (size_t)token * 4096;
#pragma unroll
    for (int ii = 0; ii < 16; ++ii) {
        int fr = ii * 256 + tid;
        float yv = (v[ii] - mean) * rstd * gamma[fr] + beta[fr];
        out[fr] = (__bf16)yv;
    }
}

// ---------------------------------------------------------------------------
// Kernel 2: QKV GEMM  (M=8192, K=4096, N=512)  blockIdx.y selects Q/K/V.
// Weights pre-transposed: WT[n][k], K contiguous -> B fragments are two
// 16B contiguous loads per lane (same as A).  Q pre-scaled by 1/8.
// Q/K out: (b,h,seq,dh).  V out TRANSPOSED: (b,h,dh,seq) for P@V fragments.
// ---------------------------------------------------------------------------
__global__ __launch_bounds__(256)
void qkv_gemm_kernel(const __bf16* __restrict__ Y,
                     const __bf16* __restrict__ WqT,
                     const __bf16* __restrict__ WkT,
                     const __bf16* __restrict__ WvT,
                     const float* __restrict__ bq,
                     const float* __restrict__ bk,
                     const float* __restrict__ bv,
                     __bf16* __restrict__ Qb,
                     __bf16* __restrict__ Kb,
                     __bf16* __restrict__ Vt) {
    const int which = blockIdx.y;
    const __bf16* WT  = (which == 0) ? WqT : ((which == 1) ? WkT : WvT);
    const float* bias = (which == 0) ? bq : ((which == 1) ? bk : bv);
    const float scale = (which == 0) ? 0.125f : 1.0f;

    const int tid  = threadIdx.x;
    const int wave = tid >> 5;
    const int lane = tid & 31;
    const int lh   = lane >> 4;
    const int ln   = lane & 15;

    const int bm = blockIdx.x >> 3;        // 0..255  (8192/32)
    const int bn = blockIdx.x & 7;         // 0..7    (512/64)
    const int row0 = bm * 32 + (wave >> 2) * 16;
    const int col0 = bn * 64 + (wave & 3) * 16;

    const __bf16* arow = Y  + (size_t)(row0 + ln) * 4096;
    const __bf16* brow = WT + (size_t)(col0 + ln) * 4096;

    f32x8 acc = {};
    for (int k0 = 0; k0 < 4096; k0 += 32) {
        bf16x16 a, bfrag;
        __builtin_prefetch(arow + k0 + 64, 0, 1);
        __builtin_prefetch(brow + k0 + 64, 0, 1);
#pragma unroll
        for (int e = 0; e < 16; ++e) {
            const int kk = (lh ? 8 : 0) + ((e & 8) ? 16 : 0) + (e & 7);
            a[e]     = arow[k0 + kk];
            bfrag[e] = brow[k0 + kk];
        }
        acc = __builtin_amdgcn_wmma_f32_16x16x32_bf16(false, a, false, bfrag,
                                                      (short)0, acc, false, false);
    }
    // epilogue: D layout row = i + 8*lh, col = col0 + ln
    const int col = col0 + ln;
    const int h   = col >> 6;
    const int dh  = col & 63;
    const float bc = bias[col];
#pragma unroll
    for (int i = 0; i < 8; ++i) {
        const int row = row0 + i + 8 * lh;
        const int b = row >> 8, q = row & 255;
        const float val = (acc[i] + bc) * scale;
        if (which == 2) {
            // V transposed: (b, h, dh, seq)
            Vt[((size_t)((b * 8 + h) * 64 + dh)) * 256 + q] = (__bf16)val;
        } else {
            __bf16* Out = (which == 0) ? Qb : Kb;
            Out[((size_t)((b * 8 + h) * 256 + q)) * 64 + dh] = (__bf16)val;
        }
    }
}

// ---------------------------------------------------------------------------
// Kernel 3: attention. One block = one (b,h) x 128-query half. 8 waves,
// each wave owns 16 queries; full 256-key score row in 16 f32 accumulators.
// V^T staged into LDS via async-to-LDS b128 (ASYNCcnt) when available.
// ---------------------------------------------------------------------------
__global__ __launch_bounds__(256)
void attn_kernel(const __bf16* __restrict__ Qb,
                 const __bf16* __restrict__ Kb,
                 const __bf16* __restrict__ Vt,
                 __bf16* __restrict__ Ob) {
    __shared__ __align__(16) __bf16 Vsh[64 * 256];   // V^T tile: 32 KB
    __shared__ __align__(16) __bf16 Psh[8][16 * 32]; //  8 KB

    const int blk  = blockIdx.x;           // 0..511
    const int bh   = blk >> 1;             // b*8 + h
    const int half = blk & 1;
    const int tid  = threadIdx.x;
    const int wave = tid >> 5, lane = tid & 31;
    const int lh   = lane >> 4, ln = lane & 15;
    const size_t base = (size_t)bh * 256 * 64;

    // ---- stage V^T (64 x 256 bf16 = 32 KB) ----
#if USE_ASYNC_LDS
    {
        const __bf16* src = Vt + base;
#pragma unroll
        for (int it = 0; it < 8; ++it) {
            const int e = (it * 256 + tid) * 8;     // 8 bf16 = 16 B per transfer
            __builtin_amdgcn_global_load_async_to_lds_b128(
                (__attribute__((address_space(1))) v4i*)(src + e),
                (__attribute__((address_space(3))) v4i*)(&Vsh[e]), 0, 0);
        }
        __builtin_amdgcn_s_wait_asynccnt(0);
    }
#else
    {
        const int4* s = (const int4*)(Vt + base);
        int4* d = (int4*)Vsh;
        for (int e = tid; e < 2048; e += 256) d[e] = s[e];
    }
#endif
    __syncthreads();

    const int q0 = half * 128 + wave * 16;

    bf16x16 qa0, qa1;
    {
        const __bf16* qrow = Qb + base + (size_t)(q0 + ln) * 64;
#pragma unroll
        for (int e = 0; e < 16; ++e) {
            const int kk = (lh ? 8 : 0) + ((e & 8) ? 16 : 0) + (e & 7);
            qa0[e] = qrow[kk];
            qa1[e] = qrow[kk + 32];
        }
    }

    f32x8 sacc[16] = {};
    for (int nt = 0; nt < 16; ++nt) {
        bf16x16 kb0, kb1;
        const __bf16* krow = Kb + base + (size_t)(nt * 16 + ln) * 64;
#pragma unroll
        for (int e = 0; e < 16; ++e) {
            const int kk = (lh ? 8 : 0) + ((e & 8) ? 16 : 0) + (e & 7);
            kb0[e] = krow[kk];
            kb1[e] = krow[kk + 32];
        }
        sacc[nt] = __builtin_amdgcn_wmma_f32_16x16x32_bf16(false, qa0, false, kb0,
                                                           (short)0, sacc[nt], false, false);
        sacc[nt] = __builtin_amdgcn_wmma_f32_16x16x32_bf16(false, qa1, false, kb1,
                                                           (short)0, sacc[nt], false, false);
    }

    // row softmax (row = i + 8*lh ; cols spread over 16 lanes x 16 accs)
    float l_i[8];
#pragma unroll
    for (int i = 0; i < 8; ++i) {
        float mx = -3.0e38f;
#pragma unroll
        for (int t = 0; t < 16; ++t) mx = fmaxf(mx, sacc[t][i]);
#pragma unroll
        for (int off = 8; off >= 1; off >>= 1) mx = fmaxf(mx, __shfl_xor(mx, off, 16));
        float sum = 0.f;
#pragma unroll
        for (int t = 0; t < 16; ++t) {
            float p = __expf(sacc[t][i] - mx);
            sacc[t][i] = p;
            sum += p;
        }
#pragma unroll
        for (int off = 8; off >= 1; off >>= 1) sum += __shfl_xor(sum, off, 16);
        l_i[i] = sum;
    }

    // O = P @ V   (V^T in LDS: element (kk, dh) at Vsh[dh*256 + key])
    f32x8 oacc[4] = {};
    for (int t2 = 0; t2 < 8; ++t2) {
#pragma unroll
        for (int i = 0; i < 8; ++i) {
            const int m = i + 8 * lh;
            Psh[wave][m * 32 + ln]      = (__bf16)sacc[2 * t2][i];
            Psh[wave][m * 32 + ln + 16] = (__bf16)sacc[2 * t2 + 1][i];
        }
        __syncthreads();
        bf16x16 pa;
#pragma unroll
        for (int e = 0; e < 16; ++e) {
            const int kk = (lh ? 8 : 0) + ((e & 8) ? 16 : 0) + (e & 7);
            pa[e] = Psh[wave][ln * 32 + kk];
        }
#pragma unroll
        for (int nt4 = 0; nt4 < 4; ++nt4) {
            bf16x16 vbf;
#pragma unroll
            for (int e = 0; e < 16; ++e) {
                const int kk = (lh ? 8 : 0) + ((e & 8) ? 16 : 0) + (e & 7);
                vbf[e] = Vsh[(nt4 * 16 + ln) * 256 + t2 * 32 + kk];
            }
            oacc[nt4] = __builtin_amdgcn_wmma_f32_16x16x32_bf16(false, pa, false, vbf,
                                                                (short)0, oacc[nt4], false, false);
        }
        __syncthreads();
    }

    const int bb = bh >> 3, hh = bh & 7;
#pragma unroll
    for (int nt4 = 0; nt4 < 4; ++nt4) {
#pragma unroll
        for (int i = 0; i < 8; ++i) {
            const int row = q0 + i + 8 * lh;
            const int dh  = nt4 * 16 + ln;
            const float val = oacc[nt4][i] / l_i[i];
            Ob[((size_t)(bb * 256 + row)) * 512 + hh * 64 + dh] = (__bf16)val;
        }
    }
}

// ---------------------------------------------------------------------------
// Kernel 4: output projection (M=8192, K=512, N=4096) + bias + residual(x)
// with fused depatchify scatter.  WoT[n][k] gives contiguous-K B fragments.
// ---------------------------------------------------------------------------
__global__ __launch_bounds__(256)
void proj_kernel(const __bf16* __restrict__ Ob,
                 const __bf16* __restrict__ WoT,
                 const float* __restrict__ bo,
                 const float* __restrict__ x,
                 float* __restrict__ out) {
    const int tid  = threadIdx.x;
    const int wave = tid >> 5, lane = tid & 31;
    const int lh   = lane >> 4, ln = lane & 15;

    const int bm = blockIdx.x >> 6;        // 0..255 (8192/32)
    const int bn = blockIdx.x & 63;        // 0..63  (4096/64)
    const int row0 = bm * 32 + (wave >> 2) * 16;
    const int col0 = bn * 64 + (wave & 3) * 16;

    const __bf16* arow = Ob  + (size_t)(row0 + ln) * 512;
    const __bf16* brow = WoT + (size_t)(col0 + ln) * 512;

    f32x8 acc = {};
    for (int k0 = 0; k0 < 512; k0 += 32) {
        bf16x16 a, bfrag;
#pragma unroll
        for (int e = 0; e < 16; ++e) {
            const int kk = (lh ? 8 : 0) + ((e & 8) ? 16 : 0) + (e & 7);
            a[e]     = arow[k0 + kk];
            bfrag[e] = brow[k0 + kk];
        }
        acc = __builtin_amdgcn_wmma_f32_16x16x32_bf16(false, a, false, bfrag,
                                                      (short)0, acc, false, false);
    }
    const int col = col0 + ln;
    const int pr = col >> 9, pc = (col >> 6) & 7, cc = col & 63;
    const float bc = bo[col];
#pragma unroll
    for (int i = 0; i < 8; ++i) {
        const int row = row0 + i + 8 * lh;
        const int b = row >> 8, tok = row & 255;
        const int pi = tok >> 4, pj = tok & 15;
        const size_t xi = ((size_t)(b * 128 + (pj * 8 + pr)) * 128 + (pi * 8 + pc)) * 64 + cc;
        out[xi] = acc[i] + bc + x[xi];
    }
}

// ---------------------------------------------------------------------------
extern "C" void kernel_launch(void* const* d_in, const int* in_sizes, int n_in,
                              void* d_out, int out_size, void* d_ws, size_t ws_size,
                              hipStream_t stream) {
    (void)in_sizes; (void)n_in; (void)out_size; (void)ws_size;
    const float* x  = (const float*)d_in[0];
    const float* g  = (const float*)d_in[1];
    const float* be = (const float*)d_in[2];
    const float* wq = (const float*)d_in[3];
    const float* bq = (const float*)d_in[4];
    const float* wk = (const float*)d_in[5];
    const float* bk = (const float*)d_in[6];
    const float* wv = (const float*)d_in[7];
    const float* bv = (const float*)d_in[8];
    const float* wo = (const float*)d_in[9];
    const float* bo = (const float*)d_in[10];
    float* out = (float*)d_out;

    // workspace layout (bf16 elements)
    const size_t NW  = 4096u * 512u;         // 2,097,152 per weight
    const size_t NY  = 8192u * 4096u;        // 33,554,432
    const size_t NQK = 8192u * 512u;         // 4,194,304
    __bf16* WqT = (__bf16*)d_ws;
    __bf16* WkT = WqT + NW;
    __bf16* WvT = WkT + NW;
    __bf16* WoT = WvT + NW;
    __bf16* Y   = WoT + NW;
    __bf16* Qb  = Y   + NY;
    __bf16* Kb  = Qb  + NQK;
    __bf16* Vt  = Kb  + NQK;
    __bf16* Ob  = Vt  + NQK;

    cvt_bf16_t_kernel<<<8192, 256, 0, stream>>>(wq, WqT, 4096, 512);
    cvt_bf16_t_kernel<<<8192, 256, 0, stream>>>(wk, WkT, 4096, 512);
    cvt_bf16_t_kernel<<<8192, 256, 0, stream>>>(wv, WvT, 4096, 512);
    cvt_bf16_t_kernel<<<8192, 256, 0, stream>>>(wo, WoT, 512, 4096);

    patchify_ln_kernel<<<8192, 256, 0, stream>>>(x, g, be, Y);

    qkv_gemm_kernel<<<dim3(2048, 3), 256, 0, stream>>>(Y, WqT, WkT, WvT,
                                                       bq, bk, bv, Qb, Kb, Vt);

    attn_kernel<<<512, 256, 0, stream>>>(Qb, Kb, Vt, Ob);

    proj_kernel<<<16384, 256, 0, stream>>>(Ob, WoT, bo, x, out);
}
